// MDVGAE_17583596110492
// MI455X (gfx1250) — matrix operand order repaired and checked by reading, not compiled, not run on hardware
//
#include <hip/hip_runtime.h>
#include <hip/hip_bf16.h>
#include <math.h>
#include <stdint.h>

// ---------------------------------------------------------------------------
// MDVGAE forward for MI455X (gfx1250, wave32, WMMA bf16 + async-LDS path).
//
// Sizes (fixed by the reference):
//   H=128, 2H=256, N=64000 nodes, E=500000 edges, B=512000 bridges.
// Output layout (flat, return order):
//   [0, 512000)        A_pred            (B,1)
//   [512000]           kl                scalar
//   [512001]           edge_attr_loss    scalar
//   [512002, ...)      bridge_attr       (B,128)
//
// Input index assumption: top-level setup_inputs() dict in insertion order,
// "params" expanded as JAX pytree leaves (dict keys sorted alphabetically):
//   0..5   cls1.{W,b} cls2.{W,b} cls3.{W,b}
//   6..13  conv1.{W1,W2,b1,b2,beta,gamma,mu,var}
//   14..21 conv2.{...}
//   22..29 conv_logstd.{...}
//   30..37 conv_mean.{...}
//   38..39 edge_pred.{W,b}
//   40 x  41 edge_index(int32)  42 edge_attr  43 bridge_idx(int32)
//   44..46 noise1..3
// ---------------------------------------------------------------------------

typedef __attribute__((ext_vector_type(16))) __bf16 bf16x16;
typedef __attribute__((ext_vector_type(8)))  float  f32x8;

namespace {
constexpr int    kH  = 128;
constexpr int    kH2 = 256;
constexpr int    kN  = 64000;
constexpr int    kE  = 500000;
constexpr int    kB  = 512000;
constexpr size_t kNH = (size_t)kN * kH;      // 8,192,000 floats
constexpr float  kBnEps = 1e-5f;
}

__device__ __forceinline__ f32x8 wmma_bf16(bf16x16 a, bf16x16 b, f32x8 c) {
  return __builtin_amdgcn_wmma_f32_16x16x32_bf16(false, a, false, b,
                                                 (short)0, c, false, false);
}

// CDNA5 async global->LDS copy (ASYNCcnt-tracked, see CDNA5 ISA).
// lds_off: byte offset into the workgroup LDS; gptr: per-lane global address.
__device__ __forceinline__ void async_copy_b128(unsigned lds_off, const void* gptr) {
  asm volatile("global_load_async_to_lds_b128 %0, %1, off"
               :: "v"(lds_off), "v"((unsigned long long)(uintptr_t)gptr)
               : "memory");
}
__device__ __forceinline__ void wait_async0() {
  asm volatile("s_wait_asynccnt 0" ::: "memory");
}

// --- Pack f32 row-major weight [K x Nout] into bf16 WMMA B-fragments. ------
// Fragment (kt,nt): 32 lanes x 16 bf16. lane -> n = nt*16+(lane&15),
// element j -> k = kt*32 + (lane>>4)*8 + j (+8 if j>=8).  One thread per (frag,lane).
__global__ __launch_bounds__(256) void pack_b_frags(
    const float* __restrict__ W, __bf16* __restrict__ out, int K, int Nout) {
  int t = blockIdx.x * blockDim.x + threadIdx.x;
  int NT = Nout >> 4;
  int total = (K >> 5) * NT * 32;
  if (t >= total) return;
  int lane = t & 31;
  int f = t >> 5;
  int nt = f % NT, kt = f / NT;
  int n = nt * 16 + (lane & 15);
  int kbase = kt * 32 + (lane >> 4) * 8;
  __bf16* dst = out + (size_t)t * 16;
#pragma unroll
  for (int j = 0; j < 16; ++j)
    dst[j] = (__bf16)W[(size_t)(kbase + j + (j >= 8 ? 8 : 0)) * Nout + n];
}

// --- agg = x (float4 copy) -------------------------------------------------
__global__ __launch_bounds__(256) void copy_f4(
    const float4* __restrict__ s, float4* __restrict__ d, int n4) {
  int i = blockIdx.x * blockDim.x + threadIdx.x;
  if (i < n4) d[i] = s[i];
}

// --- agg[dst] += relu(x[src] + edge_attr), 4 floats per thread -------------
__global__ __launch_bounds__(256) void edge_scatter(
    const float* __restrict__ x, const float* __restrict__ ea,
    const int* __restrict__ src, const int* __restrict__ dst,
    float* agg) {
  int t = blockIdx.x * blockDim.x + threadIdx.x;   // E*32 threads
  int e = t >> 5;
  if (e >= kE) return;
  int h = (t & 31) << 2;
  int s = src[e], d = dst[e];
  float4 xv = *(const float4*)(x + (size_t)s * kH + h);
  float4 ev = *(const float4*)(ea + (size_t)e * kH + h);
  float* ap = agg + (size_t)d * kH + h;
  atomicAdd(ap + 0, fmaxf(xv.x + ev.x, 0.f));
  atomicAdd(ap + 1, fmaxf(xv.y + ev.y, 0.f));
  atomicAdd(ap + 2, fmaxf(xv.z + ev.z, 0.f));
  atomicAdd(ap + 3, fmaxf(xv.w + ev.w, 0.f));
}

// --- Fused GINE MLP: out = act(BN(in @ W1 + b1)) @ W2 + b2 -----------------
// 256 threads = 8 waves, 16 rows/wave, 128 rows/block. W1p|W2p (128 KB,
// contiguous) async-staged to LDS once per block; stage-1 C tiles re-striped
// via LDS into stage-2 A fragments. BN folded to h = t*s + c.
__global__ __launch_bounds__(256) void mlp_gine(
    const float* __restrict__ in,       // [N,128] = x + sum(msg)
    const __bf16* __restrict__ Wpk,     // packed W1 frags (32768) | W2 frags (32768)
    const float* __restrict__ b1, const float* __restrict__ gamma,
    const float* __restrict__ beta, const float* __restrict__ mu,
    const float* __restrict__ var, const float* __restrict__ b2,
    float* __restrict__ out, int relu_mid) {
  __shared__ __bf16 lds_t[8][16 * 32];            // 8 KB re-stripe buffers
  extern __shared__ __bf16 lds_w[];               // 128 KB: W1p | W2p
  const int wave = threadIdx.x >> 5;
  const int lane = threadIdx.x & 31;
  const int hf   = lane >> 4;                     // lane half (K split)
  const int m_a  = lane & 15;                     // A-layout row / C-layout col
  const int rowbase = blockIdx.x * 128 + wave * 16;

  // ---- async-stage 128 KB of packed weights into LDS (ASYNCcnt path) ----
  {
    const unsigned base = (unsigned)(uintptr_t)lds_w;
#pragma unroll
    for (int i = 0; i < 32; ++i) {
      unsigned off = (i * 256 + threadIdx.x) * 16;   // bytes
      async_copy_b128(base + off, (const char*)Wpk + off);
    }
  }

  // Overlap: preload A1 fragments (row of `in`, bf16) -- K = 0..127.
  const float* rowp = in + (size_t)(rowbase + m_a) * kH;
  bf16x16 a1[4];
#pragma unroll
  for (int kt = 0; kt < 4; ++kt) {
    int k0 = kt * 32 + hf * 8;
#pragma unroll
    for (int j = 0; j < 8; ++j) a1[kt][j]     = (__bf16)rowp[k0 + j];
#pragma unroll
    for (int j = 0; j < 8; ++j) a1[kt][8 + j] = (__bf16)rowp[k0 + 16 + j];
  }

  wait_async0();
  __syncthreads();
  const __bf16* W1l = lds_w;
  const __bf16* W2l = lds_w + 32768;

  f32x8 acc2[8];
#pragma unroll
  for (int i = 0; i < 8; ++i) acc2[i] = (f32x8)(0.f);

  const int mrow0 = hf * 8;                       // C-layout row base
#pragma unroll 2
  for (int kt2 = 0; kt2 < 8; ++kt2) {             // 32 hidden cols per iter
    // ---- stage 1: t[:, kt2*32 .. +32) --------------------------------
#pragma unroll
    for (int sub = 0; sub < 2; ++sub) {
      int nt = kt2 * 2 + sub;                     // 0..15
      f32x8 c = (f32x8)(0.f);
#pragma unroll
      for (int k1 = 0; k1 < 4; ++k1) {
        bf16x16 bf = *(const bf16x16*)(W1l + ((size_t)(k1 * 16 + nt) * 32 + lane) * 16);
        c = wmma_bf16(a1[k1], bf, c);
      }
      int col = nt * 16 + m_a;
      float s  = gamma[col] * rsqrtf(var[col] + kBnEps);
      float cc = (b1[col] - mu[col]) * s + beta[col];
#pragma unroll
      for (int r = 0; r < 8; ++r) {
        float h = fmaf(c[r], s, cc);
        if (relu_mid) h = fmaxf(h, 0.f);
        lds_t[wave][(mrow0 + r) * 32 + sub * 16 + m_a] = (__bf16)h;
      }
    }
    // ---- re-stripe through LDS into stage-2 A fragment ---------------
    bf16x16 a2;
    {
      const __bf16* tp = lds_t[wave] + m_a * 32;
      int k0 = hf * 8;
#pragma unroll
      for (int j = 0; j < 8; ++j) a2[j]     = tp[k0 + j];
#pragma unroll
      for (int j = 0; j < 8; ++j) a2[8 + j] = tp[k0 + 16 + j];
    }
    // ---- stage 2 accumulate -------------------------------------------
#pragma unroll
    for (int nt2 = 0; nt2 < 8; ++nt2) {
      bf16x16 bf = *(const bf16x16*)(W2l + ((size_t)(kt2 * 8 + nt2) * 32 + lane) * 16);
      acc2[nt2] = wmma_bf16(a2, bf, acc2[nt2]);
    }
  }
  // ---- epilogue: + b2, store ------------------------------------------
#pragma unroll
  for (int nt2 = 0; nt2 < 8; ++nt2) {
    int col = nt2 * 16 + m_a;
    float bb = b2[col];
#pragma unroll
    for (int r = 0; r < 8; ++r)
      out[(size_t)(rowbase + mrow0 + r) * kH + col] = acc2[nt2][r] + bb;
  }
}

// --- z1/z2/z3 = noise*exp(logstd)+mean, fused KL reduction -----------------
// z1 aliases logstd buffer, z3 aliases mean buffer (elementwise, safe).
__global__ __launch_bounds__(256) void z_kl(
    const float* mean_in, const float* logstd_in,
    const float* __restrict__ n1, const float* __restrict__ n2,
    const float* __restrict__ n3,
    float* z1, float* z2, float* z3, float* __restrict__ kl_accum) {
  int i = blockIdx.x * blockDim.x + threadIdx.x;   // exactly kNH threads
  float m = mean_in[i], l = logstd_in[i];
  float s = __expf(l);
  z1[i] = fmaf(n1[i], s, m);
  z2[i] = fmaf(n2[i], s, m);
  z3[i] = fmaf(n3[i], s, m);
  float c = 1.f + 2.f * l - m * m - s * s;
#pragma unroll
  for (int off = 16; off; off >>= 1) c += __shfl_xor(c, off, 32);
  if ((threadIdx.x & 31) == 0)
    atomicAdd(kl_accum, c * (0.5f / ((float)kN * (float)kN)));
}

// --- A_pred: one wave per bridge, 3 classifiers fused ----------------------
__global__ __launch_bounds__(256) void a_pred(
    const float* __restrict__ z1, const float* __restrict__ z2,
    const float* __restrict__ z3,
    const int* __restrict__ bsrc, const int* __restrict__ bdst,
    const float* __restrict__ w1, const float* __restrict__ c1,
    const float* __restrict__ w2, const float* __restrict__ c2,
    const float* __restrict__ w3, const float* __restrict__ c3,
    float* __restrict__ out) {
  int g = (blockIdx.x * blockDim.x + threadIdx.x) >> 5;
  if (g >= kB) return;
  int lane = threadIdx.x & 31;
  size_t r0 = (size_t)bsrc[g] * kH, r1 = (size_t)bdst[g] * kH;
  float a1 = 0.f, a2 = 0.f, a3 = 0.f;
#pragma unroll
  for (int i = 0; i < 4; ++i) {
    int h = lane + 32 * i;
    a1 += z1[r0 + h] * w1[h] + z1[r1 + h] * w1[kH + h];
    a2 += z2[r0 + h] * w2[h] + z2[r1 + h] * w2[kH + h];
    a3 += z3[r0 + h] * w3[h] + z3[r1 + h] * w3[kH + h];
  }
#pragma unroll
  for (int off = 16; off; off >>= 1) {
    a1 += __shfl_xor(a1, off, 32);
    a2 += __shfl_xor(a2, off, 32);
    a3 += __shfl_xor(a3, off, 32);
  }
  if (lane == 0) {
    float s1 = 1.f / (1.f + __expf(-(a1 + c1[0])));
    float s2 = 1.f / (1.f + __expf(-(a2 + c2[0])));
    float s3 = 1.f / (1.f + __expf(-(a3 + c3[0])));
    out[g] = (s1 + s2 + s3) * (1.f / 3.f);
  }
}

// --- pair GEMM: rows = concat(z[i0[t]], z[i1[t]]) @ W + b ------------------
// mode 0: store out[t,128] (bridge_attr). mode 1: accumulate mean|pred-tgt|.
// 256 threads = 8 waves, 16 pairs/wave. Weights (64 KB) async-staged to LDS.
__global__ __launch_bounds__(256) void pair_gemm(
    const float* __restrict__ z,
    const int* __restrict__ i0, const int* __restrict__ i1,
    const __bf16* __restrict__ Wp,   // packed frags: KT=8, NT=8 ([256x128])
    const float* __restrict__ bvec,
    float* out, const float* __restrict__ target,
    int T, int mode, float scale) {
  extern __shared__ __bf16 lds_w[];            // 64 KB
  const int wave = threadIdx.x >> 5;
  const int lane = threadIdx.x & 31;
  const int hf   = lane >> 4;
  const int m_a  = lane & 15;
  const int tbase = (blockIdx.x * 8 + wave) * 16;

  // ---- async-stage weights (all waves participate) ----------------------
  {
    const unsigned base = (unsigned)(uintptr_t)lds_w;
#pragma unroll
    for (int i = 0; i < 16; ++i) {
      unsigned off = (i * 256 + threadIdx.x) * 16;
      async_copy_b128(base + off, (const char*)Wp + off);
    }
  }

  // Overlap: gather A fragments (pair indices clamped for tail waves).
  int p = (tbase + m_a < T) ? (tbase + m_a) : (T - 1);
  const float* rowA = z + (size_t)i0[p] * kH;
  const float* rowB = z + (size_t)i1[p] * kH;
  bf16x16 a[8];
#pragma unroll
  for (int kt = 0; kt < 8; ++kt) {             // K tile: kt<4 -> side0
    const float* rp = (kt < 4) ? rowA : rowB;
    int k0 = (kt & 3) * 32 + hf * 8;
#pragma unroll
    for (int j = 0; j < 8; ++j) a[kt][j]     = (__bf16)rp[k0 + j];
#pragma unroll
    for (int j = 0; j < 8; ++j) a[kt][8 + j] = (__bf16)rp[k0 + 16 + j];
  }

  wait_async0();
  __syncthreads();
  if (tbase >= T) return;                      // wave-uniform: EXEC stays full

  const int mrow0 = hf * 8;
  float lsum = 0.f;
#pragma unroll
  for (int nt = 0; nt < 8; ++nt) {
    f32x8 c = (f32x8)(0.f);
#pragma unroll
    for (int kt = 0; kt < 8; ++kt) {
      bf16x16 bf = *(const bf16x16*)(lds_w + ((size_t)(kt * 8 + nt) * 32 + lane) * 16);
      c = wmma_bf16(a[kt], bf, c);
    }
    int col = nt * 16 + m_a;
    float bb = bvec[col];
    if (mode == 0) {
#pragma unroll
      for (int r = 0; r < 8; ++r)
        out[(size_t)(tbase + mrow0 + r) * kH + col] = c[r] + bb;
    } else {
#pragma unroll
      for (int r = 0; r < 8; ++r) {
        size_t row = (size_t)(tbase + mrow0 + r);
        lsum += fabsf(c[r] + bb - target[row * kH + col]);
      }
    }
  }
  if (mode == 1) {
#pragma unroll
    for (int off = 16; off; off >>= 1) lsum += __shfl_xor(lsum, off, 32);
    if (lane == 0) atomicAdd(out, lsum * scale);
  }
}

__global__ void init_scalars(float* out) { out[kB] = 0.f; out[kB + 1] = 0.f; }

// ---------------------------------------------------------------------------
struct ConvP { const float *W1, *W2, *b1, *b2, *beta, *gamma, *mu, *var; };
static ConvP get_conv(void* const* d_in, int base) {
  return { (const float*)d_in[base + 0], (const float*)d_in[base + 1],
           (const float*)d_in[base + 2], (const float*)d_in[base + 3],
           (const float*)d_in[base + 4], (const float*)d_in[base + 5],
           (const float*)d_in[base + 6], (const float*)d_in[base + 7] };
}

extern "C" void kernel_launch(void* const* d_in, const int* in_sizes, int n_in,
                              void* d_out, int out_size, void* d_ws, size_t ws_size,
                              hipStream_t stream) {
  (void)in_sizes; (void)n_in; (void)out_size; (void)ws_size;

  // --- input pointers (see index map at top) -------------------------------
  const float* cls1W = (const float*)d_in[0];  const float* cls1b = (const float*)d_in[1];
  const float* cls2W = (const float*)d_in[2];  const float* cls2b = (const float*)d_in[3];
  const float* cls3W = (const float*)d_in[4];  const float* cls3b = (const float*)d_in[5];
  ConvP c1  = get_conv(d_in, 6);
  ConvP c2  = get_conv(d_in, 14);
  ConvP cls = get_conv(d_in, 22);   // conv_logstd
  ConvP cme = get_conv(d_in, 30);   // conv_mean
  const float* epW = (const float*)d_in[38];
  const float* epb = (const float*)d_in[39];
  const float* x   = (const float*)d_in[40];
  const int*   esrc = (const int*)d_in[41];
  const int*   edst = esrc + kE;
  const float* eattr = (const float*)d_in[42];
  const int*   bsrc = (const int*)d_in[43];
  const int*   bdst = bsrc + kB;
  const float* n1 = (const float*)d_in[44];
  const float* n2 = (const float*)d_in[45];
  const float* n3 = (const float*)d_in[46];
  float* out = (float*)d_out;

  // --- workspace layout ----------------------------------------------------
  float* agg    = (float*)d_ws;
  float* h1     = agg    + kNH;      // later: z2
  float* h2     = h1     + kNH;
  float* meanb  = h2     + kNH;      // later: z3
  float* lstdb  = meanb  + kNH;      // later: z1
  __bf16* wpk   = (__bf16*)(lstdb + kNH);
  __bf16* Wpk[4];                    // per conv: W1 frags | W2 frags (contig)
  const ConvP* convs[4] = { &c1, &c2, &cme, &cls };
  for (int i = 0; i < 4; ++i) Wpk[i] = wpk + (size_t)i * 65536;
  __bf16* epWp = wpk + 4 * 65536;

  // --- 1. pack weights to bf16 WMMA B-fragment layout ----------------------
  for (int i = 0; i < 4; ++i) {
    pack_b_frags<<<8, 256, 0, stream>>>(convs[i]->W1, Wpk[i], kH, kH2);
    pack_b_frags<<<8, 256, 0, stream>>>(convs[i]->W2, Wpk[i] + 32768, kH2, kH);
  }
  pack_b_frags<<<8, 256, 0, stream>>>(epW, epWp, kH2, kH);
  init_scalars<<<1, 1, 0, stream>>>(out);

  const int copyBlocks    = (int)(kNH / 4 / 256);   // 8000
  const int scatterBlocks = kE * 32 / 256;          // 62500
  const int mlpBlocks     = kN / 128;               // 500
  const size_t mlpLds     = 131072;                 // 128 KB dynamic
  const size_t pairLds    = 65536;                  // 64 KB dynamic

  // --- 2. conv1: x -> h1 ---------------------------------------------------
  copy_f4<<<copyBlocks, 256, 0, stream>>>((const float4*)x, (float4*)agg, (int)(kNH / 4));
  edge_scatter<<<scatterBlocks, 256, 0, stream>>>(x, eattr, esrc, edst, agg);
  mlp_gine<<<mlpBlocks, 256, mlpLds, stream>>>(agg, Wpk[0], c1.b1, c1.gamma,
      c1.beta, c1.mu, c1.var, c1.b2, h1, 1);

  // --- 3. conv2: h1 -> h2 --------------------------------------------------
  copy_f4<<<copyBlocks, 256, 0, stream>>>((const float4*)h1, (float4*)agg, (int)(kNH / 4));
  edge_scatter<<<scatterBlocks, 256, 0, stream>>>(h1, eattr, esrc, edst, agg);
  mlp_gine<<<mlpBlocks, 256, mlpLds, stream>>>(agg, Wpk[1], c2.b1, c2.gamma,
      c2.beta, c2.mu, c2.var, c2.b2, h2, 1);

  // --- 4. conv_mean / conv_logstd share agg(h2) ----------------------------
  copy_f4<<<copyBlocks, 256, 0, stream>>>((const float4*)h2, (float4*)agg, (int)(kNH / 4));
  edge_scatter<<<scatterBlocks, 256, 0, stream>>>(h2, eattr, esrc, edst, agg);
  mlp_gine<<<mlpBlocks, 256, mlpLds, stream>>>(agg, Wpk[2], cme.b1, cme.gamma,
      cme.beta, cme.mu, cme.var, cme.b2, meanb, 0);
  mlp_gine<<<mlpBlocks, 256, mlpLds, stream>>>(agg, Wpk[3], cls.b1, cls.gamma,
      cls.beta, cls.mu, cls.var, cls.b2, lstdb, 0);

  // --- 5. z's + KL (z1 -> lstdb, z2 -> h1, z3 -> meanb, in place) ----------
  z_kl<<<(int)(kNH / 256), 256, 0, stream>>>(meanb, lstdb, n1, n2, n3,
      lstdb, h1, meanb, out + kB);
  float* z1 = lstdb; float* z2 = h1; float* z3 = meanb;

  // --- 6. A_pred -----------------------------------------------------------
  a_pred<<<kB * 32 / 256, 256, 0, stream>>>(z1, z2, z3, bsrc, bdst,
      cls1W, cls1b, cls2W, cls2b, cls3W, cls3b, out);

  // --- 7. edge_attr_loss (fused reduction, never materializes [E,H]) -------
  {
    int blocks = (kE / 16 + 7) / 8;   // 3907
    pair_gemm<<<blocks, 256, pairLds, stream>>>(z1, esrc, edst, epWp, epb,
        out + kB + 1, eattr, kE, 1, 1.0f / ((float)kE * (float)kH));
  }
  // --- 8. bridge_attr ------------------------------------------------------
  {
    int blocks = kB / 128;            // 4000
    pair_gemm<<<blocks, 256, pairLds, stream>>>(z1, bsrc, bdst, epWp, epb,
        out + kB + 2, nullptr, kB, 0, 0.f);
  }
}
